// MultiHeadAttention_6854767804478
// MI455X (gfx1250) — compile-verified, matrix operand order checked by reading, not run on hardware
//
#include <hip/hip_runtime.h>
#include <hip/hip_bf16.h>

#define TT 4096
#define EE 512
#define HH 8
#define DHH 64
#define MM (2*TT)       // 8192 rows of x
#define LDS_STRIDE 40   // 32 k-elems + 8 pad (80B rows: 16B aligned, conflict-free)
#define KV_STRIDE 72    // 64 elems + 8 pad (144B rows: 16B aligned, conflict-free)

typedef __attribute__((ext_vector_type(16))) __bf16 bf16x16;
typedef __attribute__((ext_vector_type(8)))  float  f32x8;
typedef __attribute__((ext_vector_type(4)))  unsigned int u32x4;
typedef __attribute__((ext_vector_type(8)))  int i32x8;
typedef __attribute__((ext_vector_type(4)))  int i32x4;

#if __has_builtin(__builtin_amdgcn_tensor_load_to_lds) && __has_builtin(__builtin_amdgcn_s_wait_tensorcnt)
#define USE_TDM 1
#else
#define USE_TDM 0
#endif

union F8   { f32x8 v; float f[8]; };
union FragU { struct { uint4 lo, hi; } u; bf16x16 f; };

__device__ inline bf16x16 frag_ld(const __bf16* p) {
  FragU x;
  x.u.lo = *(const uint4*)(p);
  x.u.hi = *(const uint4*)(p + 8);
  return x.f;
}

__device__ inline f32x8 wmma_bf16(bf16x16 a, bf16x16 b, f32x8 c) {
  // (neg_a, A, neg_b, B, c_mod, C, reuse_a, reuse_b)
  return __builtin_amdgcn_wmma_f32_16x16x32_bf16(false, a, false, b, (short)0, c, false, false);
}

__device__ inline void cvt_store4(__bf16* dst, float4 v) {
  union { __bf16 h[4]; uint2 u; } p;
  p.h[0] = (__bf16)v.x; p.h[1] = (__bf16)v.y;
  p.h[2] = (__bf16)v.z; p.h[3] = (__bf16)v.w;
  *(uint2*)dst = p.u;
}

#if USE_TDM
__device__ inline unsigned lds_off(const void* p) {
  // generic -> addrspace(3) -> 32-bit LDS byte offset
  return (unsigned)(__SIZE_TYPE__)(const __attribute__((address_space(3))) char*)p;
}

// Issue a TDM load of a 64x64 bf16 tile (row stride `stride_elems` in global)
// into LDS at byte offset lds_base, with LDS padding of 4 DWORDs every 32
// DWORDs (-> 72-element rows).
__device__ inline void tdm_issue_tile(const __bf16* g, unsigned lds_base,
                                      unsigned stride_elems) {
  unsigned long long ga = (unsigned long long)(__SIZE_TYPE__)g;
  u32x4 g0;
  g0[0] = 1u;                                       // count=1 (valid descriptor)
  g0[1] = lds_base;                                 // lds_addr
  g0[2] = (unsigned)ga;                             // global_addr[31:0]
  g0[3] = (unsigned)((ga >> 32) & 0x01FFFFFFu) | (2u << 30); // addr[56:32] | type=2
  i32x8 g1;
  g1[0] = (int)((1u << 16)      // data_size = 2 bytes
              | (1u << 20)      // pad_enable
              | (4u << 22)      // pad_interval: 32 DWORDs
              | (3u << 25));    // pad_amount: 4 DWORDs (8 bf16)
  g1[1] = (int)(64u << 16);     // tensor_dim0 = 64 (bits 79:48, low half)
  g1[2] = (int)(64u << 16);     // tensor_dim1 = 64 (bits 111:80, low half)
  g1[3] = (int)(64u << 16);     // tile_dim0 = 64 (bits 127:112)
  g1[4] = 64;                   // tile_dim1 = 64 (bits 143:128)
  g1[5] = (int)stride_elems;    // tensor_dim0_stride (bits 207:160, low 32)
  g1[6] = 0;
  g1[7] = 0;
  i32x4 gz = {0, 0, 0, 0};
  i32x8 gz8 = {0, 0, 0, 0, 0, 0, 0, 0};
  // 6-arg form (clang-23 / therock headers): g0, g1, g2, g3, extra group, cpol
  __builtin_amdgcn_tensor_load_to_lds(g0, g1, gz, gz, gz8, 0);
}
#endif

// ---------------------------------------------------------------------------
// Kernel 1: qkv = x @ w_qkv.T  (M=8192, N=1536, K=512), scatter into
//   Q[b,h,t,d], K[b,h,t,d], Vt[b,h,d,t]  (bf16)
// n decomposes as n = d*24 + k3*8 + hh  (reshape(b,t,dh,3,h))
// ---------------------------------------------------------------------------
__global__ __launch_bounds__(256) void qkv_proj_kernel(
    const float* __restrict__ x, const float* __restrict__ w,
    __bf16* __restrict__ Qw, __bf16* __restrict__ Kw, __bf16* __restrict__ Vtw)
{
  __shared__ __bf16 As[128*LDS_STRIDE];
  __shared__ __bf16 Bs[128*LDS_STRIDE];
  const int tid = threadIdx.x, lane = tid & 31, wid = tid >> 5;
  const int wm = wid & 3, wn = wid >> 2;
  const int m0 = blockIdx.x * 128, n0 = blockIdx.y * 128;
  f32x8 acc[2][4] = {};

  for (int kc = 0; kc < EE; kc += 32) {
    #pragma unroll
    for (int q = 0; q < 4; ++q) {
      int e = q*256 + tid;
      int row = e >> 3, kq = e & 7;
      float4 av = *(const float4*)(x + (size_t)(m0+row)*EE + kc + kq*4);
      cvt_store4(&As[row*LDS_STRIDE + kq*4], av);
      float4 bv = *(const float4*)(w + (size_t)(n0+row)*EE + kc + kq*4);
      cvt_store4(&Bs[row*LDS_STRIDE + kq*4], bv);
    }
    __syncthreads();
    bf16x16 af[2], bfr[4];
    #pragma unroll
    for (int mt = 0; mt < 2; ++mt) {
      int r0 = wm*32 + mt*16 + (lane & 15);
      af[mt] = frag_ld(&As[r0*LDS_STRIDE + (lane>>4)*16]);
    }
    #pragma unroll
    for (int nt = 0; nt < 4; ++nt) {
      int c0 = wn*64 + nt*16 + (lane & 15);
      bfr[nt] = frag_ld(&Bs[c0*LDS_STRIDE + (lane>>4)*16]);
    }
    #pragma unroll
    for (int mt = 0; mt < 2; ++mt)
      #pragma unroll
      for (int nt = 0; nt < 4; ++nt)
        acc[mt][nt] = wmma_bf16(af[mt], bfr[nt], acc[mt][nt]);
    __syncthreads();
  }

  const int b = m0 >> 12;   // 128 | 4096, so whole block in one batch
  #pragma unroll
  for (int mt = 0; mt < 2; ++mt) {
    #pragma unroll
    for (int nt = 0; nt < 4; ++nt) {
      int n = n0 + wn*64 + nt*16 + (lane & 15);
      int d = n / 24, rem = n % 24;
      int k3 = rem >> 3, hh = rem & 7;
      F8 u; u.v = acc[mt][nt];
      #pragma unroll
      for (int r = 0; r < 8; ++r) {
        int m = m0 + wm*32 + mt*16 + r + ((lane>>4)<<3);
        int t = m & (TT-1);
        __bf16 hv = (__bf16)u.f[r];
        if (k3 == 0)      Qw [(((size_t)b*HH + hh)*TT + t)*DHH + d] = hv;
        else if (k3 == 1) Kw [(((size_t)b*HH + hh)*TT + t)*DHH + d] = hv;
        else              Vtw[(((size_t)b*HH + hh)*DHH + d)*TT + t] = hv;
      }
    }
  }
}

// ---------------------------------------------------------------------------
// Kernel 2: flash attention per (b,h). Block = 4 waves x 16 Q rows = 64 rows.
// KV tiles of 64 staged in LDS by the Tensor Data Mover (double-buffered);
// falls back to cooperative vector copies when the TDM builtin is absent.
// O written bf16 as [b, t, h*64+d].
// ---------------------------------------------------------------------------
__global__ __launch_bounds__(128) void flash_attn_kernel(
    const __bf16* __restrict__ Qw, const __bf16* __restrict__ Kw,
    const __bf16* __restrict__ Vtw, __bf16* __restrict__ Ow)
{
  __shared__ __bf16 Ks[2][64*KV_STRIDE];
  __shared__ __bf16 Vs[2][64*KV_STRIDE];
  __shared__ __bf16 Pl[4*16*KV_STRIDE];
  const int tid = threadIdx.x, lane = tid & 31, wid = tid >> 5;
  const int bh = blockIdx.y;
  const int b = bh >> 3, h = bh & 7;
  const __bf16* Qb  = Qw  + (size_t)bh*TT*DHH;
  const __bf16* Kb  = Kw  + (size_t)bh*TT*DHH;
  const __bf16* Vtb = Vtw + (size_t)bh*DHH*TT;
  const int rowbase = blockIdx.x*64 + wid*16;
  const float sc2 = 0.125f * 1.44269504f;  // dh^-0.5 * log2(e)
  __bf16* Pw = &Pl[wid*16*KV_STRIDE];
  const int half8 = (lane >> 4) << 3;

  bf16x16 qa[2];
  #pragma unroll
  for (int dc = 0; dc < 2; ++dc)
    qa[dc] = frag_ld(Qb + (size_t)(rowbase + (lane&15))*DHH + dc*32 + (lane>>4)*16);

  float mrow[8], lrow[8];
  f32x8 o[4] = {};
  #pragma unroll
  for (int r = 0; r < 8; ++r) { mrow[r] = -3.0e38f; lrow[r] = 0.f; }

#if USE_TDM
  if (wid == 0) {  // prologue DMA: tile 0 into buffer 0
    tdm_issue_tile(Kb,  lds_off(&Ks[0][0]), DHH);
    tdm_issue_tile(Vtb, lds_off(&Vs[0][0]), TT);
  }
#endif

  for (int j0 = 0; j0 < TT; j0 += 64) {
#if USE_TDM
    const int buf = (j0 >> 6) & 1;
    if (wid == 0) {
      if (j0 + 64 < TT) {
        // overlap: kick next tile's DMA, then wait only for the current pair
        tdm_issue_tile(Kb  + (size_t)(j0+64)*DHH, lds_off(&Ks[buf^1][0]), DHH);
        tdm_issue_tile(Vtb + (size_t)(j0+64),     lds_off(&Vs[buf^1][0]), TT);
        __builtin_amdgcn_s_wait_tensorcnt(2);
      } else {
        __builtin_amdgcn_s_wait_tensorcnt(0);
      }
    }
    __syncthreads();
    const __bf16* Kcur = &Ks[buf][0];
    const __bf16* Vcur = &Vs[buf][0];
#else
    // cooperative copy: 64x64 bf16 K tile + Vt tile, 16B chunks
    #pragma unroll
    for (int c = 0; c < 4; ++c) {
      int e = tid + c*128;
      int row = e >> 3, c8 = e & 7;
      *(uint4*)&Ks[0][row*KV_STRIDE + c8*8] =
          *(const uint4*)(Kb + (size_t)(j0+row)*DHH + c8*8);
      *(uint4*)&Vs[0][row*KV_STRIDE + c8*8] =
          *(const uint4*)(Vtb + (size_t)row*TT + j0 + c8*8);
    }
    __syncthreads();
    const __bf16* Kcur = &Ks[0][0];
    const __bf16* Vcur = &Vs[0][0];
#endif

    // S = Q K^T for this 16x64 strip (K fragments from LDS)
    f32x8 s[4] = {};
    #pragma unroll
    for (int nt = 0; nt < 4; ++nt)
      #pragma unroll
      for (int dc = 0; dc < 2; ++dc) {
        bf16x16 kf = frag_ld(Kcur + (nt*16 + (lane&15))*KV_STRIDE + dc*32 + (lane>>4)*16);
        s[nt] = wmma_bf16(qa[dc], kf, s[nt]);
      }
    F8 us[4];
    #pragma unroll
    for (int nt = 0; nt < 4; ++nt) us[nt].v = s[nt];

    // running row max (row lives across one 16-lane half)
    float alpha[8];
    #pragma unroll
    for (int r = 0; r < 8; ++r) {
      float v = fmaxf(fmaxf(us[0].f[r], us[1].f[r]), fmaxf(us[2].f[r], us[3].f[r]));
      v = fmaxf(v, __shfl_xor(v, 1, 32));
      v = fmaxf(v, __shfl_xor(v, 2, 32));
      v = fmaxf(v, __shfl_xor(v, 4, 32));
      v = fmaxf(v, __shfl_xor(v, 8, 32));
      float mnew = fmaxf(mrow[r], v);
      alpha[r] = exp2f((mrow[r] - mnew) * sc2);
      mrow[r] = mnew;
    }
    // p = exp(scale*(s-m)); write bf16 P to LDS (C-layout -> row-major strip)
    float ps[8];
    #pragma unroll
    for (int r = 0; r < 8; ++r) ps[r] = 0.f;
    #pragma unroll
    for (int nt = 0; nt < 4; ++nt)
      #pragma unroll
      for (int r = 0; r < 8; ++r) {
        float p = exp2f((us[nt].f[r] - mrow[r]) * sc2);
        ps[r] += p;
        Pw[(r + half8)*KV_STRIDE + nt*16 + (lane & 15)] = (__bf16)p;
      }
    #pragma unroll
    for (int r = 0; r < 8; ++r) {
      float v = ps[r];
      v += __shfl_xor(v, 1, 32);
      v += __shfl_xor(v, 2, 32);
      v += __shfl_xor(v, 4, 32);
      v += __shfl_xor(v, 8, 32);
      lrow[r] = lrow[r]*alpha[r] + v;
    }
    #pragma unroll
    for (int dt = 0; dt < 4; ++dt) {
      F8 uo; uo.v = o[dt];
      #pragma unroll
      for (int r = 0; r < 8; ++r) uo.f[r] *= alpha[r];
      o[dt] = uo.v;
    }
    asm volatile("s_wait_dscnt 0" ::: "memory"); // P stores visible before reload
    // O += P V  (A = P from LDS, B = Vt tile rows from LDS)
    #pragma unroll
    for (int jc = 0; jc < 2; ++jc) {
      bf16x16 pf = frag_ld(Pw + (lane&15)*KV_STRIDE + jc*32 + (lane>>4)*16);
      #pragma unroll
      for (int dt = 0; dt < 4; ++dt) {
        bf16x16 vf = frag_ld(Vcur + (dt*16 + (lane&15))*KV_STRIDE + jc*32 + (lane>>4)*16);
        o[dt] = wmma_bf16(pf, vf, o[dt]);
      }
    }
    __syncthreads();  // all reads of this buffer done before it is re-filled
  }

  float inv[8];
  #pragma unroll
  for (int r = 0; r < 8; ++r) inv[r] = 1.0f / lrow[r];
  #pragma unroll
  for (int dt = 0; dt < 4; ++dt) {
    F8 uo; uo.v = o[dt];
    #pragma unroll
    for (int r = 0; r < 8; ++r) {
      int t = rowbase + r + half8;
      int e = h*DHH + dt*16 + (lane & 15);
      Ow[((size_t)b*TT + t)*EE + e] = (__bf16)(uo.f[r]*inv[r]);
    }
  }
}

// ---------------------------------------------------------------------------
// Kernel 3: out = O @ w_out.T  (M=8192, N=512, K=512), fp32 output
// ---------------------------------------------------------------------------
__global__ __launch_bounds__(256) void out_proj_kernel(
    const __bf16* __restrict__ Ow, const float* __restrict__ w,
    float* __restrict__ out)
{
  __shared__ __bf16 As[128*LDS_STRIDE];
  __shared__ __bf16 Bs[128*LDS_STRIDE];
  const int tid = threadIdx.x, lane = tid & 31, wid = tid >> 5;
  const int wm = wid & 3, wn = wid >> 2;
  const int m0 = blockIdx.x * 128, n0 = blockIdx.y * 128;
  f32x8 acc[2][4] = {};

  for (int kc = 0; kc < EE; kc += 32) {
    {
      int row = tid >> 1, hf = tid & 1;
      const __bf16* src = Ow + (size_t)(m0+row)*EE + kc + hf*16;
      uint4 lo = *(const uint4*)src;
      uint4 hi = *(const uint4*)(src + 8);
      __bf16* dst = &As[row*LDS_STRIDE + hf*16];
      *(uint4*)dst = lo;
      *(uint4*)(dst + 8) = hi;
    }
    #pragma unroll
    for (int q = 0; q < 4; ++q) {
      int e = q*256 + tid;
      int row = e >> 3, kq = e & 7;
      float4 bv = *(const float4*)(w + (size_t)(n0+row)*EE + kc + kq*4);
      cvt_store4(&Bs[row*LDS_STRIDE + kq*4], bv);
    }
    __syncthreads();
    bf16x16 af[2], bfr[4];
    #pragma unroll
    for (int mt = 0; mt < 2; ++mt) {
      int r0 = wm*32 + mt*16 + (lane & 15);
      af[mt] = frag_ld(&As[r0*LDS_STRIDE + (lane>>4)*16]);
    }
    #pragma unroll
    for (int nt = 0; nt < 4; ++nt) {
      int c0 = wn*64 + nt*16 + (lane & 15);
      bfr[nt] = frag_ld(&Bs[c0*LDS_STRIDE + (lane>>4)*16]);
    }
    #pragma unroll
    for (int mt = 0; mt < 2; ++mt)
      #pragma unroll
      for (int nt = 0; nt < 4; ++nt)
        acc[mt][nt] = wmma_bf16(af[mt], bfr[nt], acc[mt][nt]);
    __syncthreads();
  }

  #pragma unroll
  for (int mt = 0; mt < 2; ++mt)
    #pragma unroll
    for (int nt = 0; nt < 4; ++nt) {
      int n = n0 + wn*64 + nt*16 + (lane & 15);
      F8 u; u.v = acc[mt][nt];
      #pragma unroll
      for (int r = 0; r < 8; ++r) {
        int m = m0 + wm*32 + mt*16 + r + ((lane>>4)<<3);
        out[(size_t)m*EE + n] = u.f[r];
      }
    }
}

// ---------------------------------------------------------------------------
extern "C" void kernel_launch(void* const* d_in, const int* in_sizes, int n_in,
                              void* d_out, int out_size, void* d_ws, size_t ws_size,
                              hipStream_t stream) {
  const float* x     = (const float*)d_in[0];
  const float* w_qkv = (const float*)d_in[1];
  const float* w_out = (const float*)d_in[2];
  float* out = (float*)d_out;

  char* ws = (char*)d_ws;
  const size_t seg = (size_t)2*HH*TT*DHH*sizeof(__bf16);  // 8 MB each
  __bf16* Qw  = (__bf16*)(ws);
  __bf16* Kw  = (__bf16*)(ws + seg);
  __bf16* Vtw = (__bf16*)(ws + 2*seg);
  __bf16* Ow  = (__bf16*)(ws + 3*seg);

  qkv_proj_kernel<<<dim3(MM/128, 1536/128), 256, 0, stream>>>(x, w_qkv, Qw, Kw, Vtw);
  flash_attn_kernel<<<dim3(TT/64, 2*HH), 128, 0, stream>>>(Qw, Kw, Vtw, Ow);
  out_proj_kernel<<<dim3(MM/128, EE/128), 256, 0, stream>>>(Ow, w_out, out);
}
